// GroupedQueryAttention_60644938219777
// MI455X (gfx1250) — compile-verified, matrix operand order checked by reading, not run on hardware
//
#include <hip/hip_runtime.h>
#include <hip/hip_bf16.h>

typedef __attribute__((ext_vector_type(16))) __bf16 bf16x16;
typedef __attribute__((ext_vector_type(8)))  float  floatx8;

// Load a 16-bit WMMA A/B fragment from row-major LDS: two 16B chunks.
// Per ISA 7.12.2: lane<16 holds K {kb..kb+7, kb+16..kb+23} with kb = (lane>=16)*8.
__device__ __forceinline__ bf16x16 frag_ld(const __bf16* p0, const __bf16* p1) {
    union { bf16x16 v; uint4 u[2]; } f;
    f.u[0] = *reinterpret_cast<const uint4*>(p0);
    f.u[1] = *reinterpret_cast<const uint4*>(p1);
    return f.v;
}

__device__ __forceinline__ floatx8 wmma_bf16(bf16x16 a, bf16x16 b, floatx8 c) {
    return __builtin_amdgcn_wmma_f32_16x16x32_bf16(false, a, false, b, (short)0, c,
                                                   false, false);
}

// Async global->LDS 16B copy (ASYNCcnt-tracked, bypasses VGPRs).
__device__ __forceinline__ void async_ld_b128(void* lds_ptr, const void* gptr) {
    unsigned l = (unsigned)(uintptr_t)lds_ptr;   // generic addr[31:0] == LDS offset
    asm volatile("global_load_async_to_lds_b128 %0, %1, off"
                 :: "v"(l), "v"(gptr) : "memory");
}
__device__ __forceinline__ void wait_async() {
    asm volatile("s_wait_asynccnt 0" ::: "memory");
}

// ---------------------------------------------------------------- casts
__global__ __launch_bounds__(256) void cast_f32_to_bf16(const float* __restrict__ in,
                                                        __bf16* __restrict__ out, int n) {
    int i = (blockIdx.x * blockDim.x + threadIdx.x) * 4;
    if (i < n) {
        float4 f = *reinterpret_cast<const float4*>(in + i);
        out[i + 0] = (__bf16)f.x;
        out[i + 1] = (__bf16)f.y;
        out[i + 2] = (__bf16)f.z;
        out[i + 3] = (__bf16)f.w;
    }
}

// out[n][k] = (bf16) in[k][n] : tiled 32x32 transpose-cast (weights -> W^T bf16)
__global__ __launch_bounds__(256) void transpose_cast_kernel(
    const float* __restrict__ in, __bf16* __restrict__ out, int K, int N) {
    __shared__ float tile[32][33];
    const int k0 = blockIdx.y * 32, n0 = blockIdx.x * 32;
    const int tx = threadIdx.x & 31, ty = threadIdx.x >> 5;
    for (int r = ty; r < 32; r += 8)
        tile[r][tx] = in[(size_t)(k0 + r) * N + n0 + tx];
    __syncthreads();
    for (int r = ty; r < 32; r += 8)
        out[(size_t)(n0 + r) * K + k0 + tx] = (__bf16)tile[tx][r];
}

// ---------------------------------------------------------------- bf16 GEMM
// C = (A[M,K] @ B^T[N,K]^T + bias) * alpha ; A, Bt row-major bf16.
// TRANS=false: C row-major [M,N] (OUT = bf16 or f32)
// TRANS=true : C transposed, [N,M] bf16, packed 16B stores.
#define GBM 128
#define GBN 128
#define GBK 32
#define GLD 40   // padded LDS stride (multiple of 8 bf16 -> 16B-aligned chunks)

template <typename OUT, bool TRANS>
__global__ __launch_bounds__(256) void gemm_bf16_kernel(
    const __bf16* __restrict__ A, const __bf16* __restrict__ Bt,
    const float* __restrict__ bias, OUT* __restrict__ C,
    int M, int N, int K, float alpha) {
    __shared__ __align__(16) __bf16 sA[2][GBM][GLD];   // [buf][m][k]
    __shared__ __align__(16) __bf16 sB[2][GBN][GLD];   // [buf][n][k]

    const int tid  = threadIdx.x;
    const int lane = tid & 31;
    const int wave = tid >> 5;
    const int wr   = wave >> 1;     // 0..3 -> 32-row slab
    const int wc   = wave & 1;      // 0..1 -> 64-col slab
    const int m0   = blockIdx.y * GBM;
    const int n0   = blockIdx.x * GBN;

    const int half = lane >> 4;
    const int ml   = lane & 15;
    const int kb   = half * 8;

    auto stage = [&](int buf, int k0) {
        for (int i = tid; i < (GBM * GBK) / 8; i += 256) {
            int r = i >> 2, c = (i & 3) * 8;
            async_ld_b128(&sA[buf][r][c], &A[(size_t)(m0 + r) * K + k0 + c]);
        }
        for (int i = tid; i < (GBN * GBK) / 8; i += 256) {
            int r = i >> 2, c = (i & 3) * 8;
            async_ld_b128(&sB[buf][r][c], &Bt[(size_t)(n0 + r) * K + k0 + c]);
        }
    };

    floatx8 acc[2][4];
#pragma unroll
    for (int rt = 0; rt < 2; ++rt)
#pragma unroll
        for (int ct = 0; ct < 4; ++ct) acc[rt][ct] = (floatx8){};

    int buf = 0;
    stage(0, 0);
    for (int k0 = 0; k0 < K; k0 += GBK) {
        wait_async();
        __syncthreads();                       // current buffer ready everywhere
        if (k0 + GBK < K) stage(buf ^ 1, k0 + GBK);   // prefetch under compute

        bf16x16 af[2], bf[4];
#pragma unroll
        for (int rt = 0; rt < 2; ++rt) {
            int r = wr * 32 + rt * 16 + ml;
            af[rt] = frag_ld(&sA[buf][r][kb], &sA[buf][r][kb + 16]);
        }
#pragma unroll
        for (int ct = 0; ct < 4; ++ct) {
            int n = wc * 64 + ct * 16 + ml;
            bf[ct] = frag_ld(&sB[buf][n][kb], &sB[buf][n][kb + 16]);
        }
#pragma unroll
        for (int rt = 0; rt < 2; ++rt)
#pragma unroll
            for (int ct = 0; ct < 4; ++ct)
                acc[rt][ct] = wmma_bf16(af[rt], bf[ct], acc[rt][ct]);
        buf ^= 1;
    }

    // epilogue: bias then alpha
#pragma unroll
    for (int rt = 0; rt < 2; ++rt)
#pragma unroll
        for (int ct = 0; ct < 4; ++ct) {
            int ng = n0 + wc * 64 + ct * 16 + ml;
            float bv = bias[ng];
            if constexpr (TRANS) {
                // fixed lane holds 8 consecutive rows -> one packed 16B store
                union { uint4 u; __bf16 h[8]; } pk;
                int mg0 = m0 + wr * 32 + rt * 16 + 8 * half;
#pragma unroll
                for (int i = 0; i < 8; ++i)
                    pk.h[i] = (__bf16)((acc[rt][ct][i] + bv) * alpha);
                *reinterpret_cast<uint4*>(&C[(size_t)ng * M + mg0]) = pk.u;
            } else {
#pragma unroll
                for (int i = 0; i < 8; ++i) {
                    int mg = m0 + wr * 32 + rt * 16 + i + 8 * half;
                    C[(size_t)mg * N + ng] = (OUT)((acc[rt][ct][i] + bv) * alpha);
                }
            }
        }
}

// ---------------------------------------------------------------- flash attention
// grid = (T/64, B*H), 128 threads (4 waves). Wave w owns Q rows [w*16, w*16+16).
// V is provided pre-transposed: Vt[KV][B*T].
__global__ __launch_bounds__(128) void gqa_flash_kernel(
    const __bf16* __restrict__ Q, const __bf16* __restrict__ Kb,
    const __bf16* __restrict__ Vt, __bf16* __restrict__ O) {
    const int T = 2048, D = 4096, HD = 128, REP = 4, KV = 1024, H = 32;

    __shared__ __align__(16) __bf16 sQ[64][136];        // [q][k]
    __shared__ __align__(16) __bf16 sK[2][64][136];     // [buf][key][k]
    __shared__ __align__(16) __bf16 sVt[2][128][72];    // [buf][n][key]
    __shared__ __align__(16) __bf16 sP[4][16][72];      // per-wave P

    const int tid = threadIdx.x, lane = tid & 31, wave = tid >> 5;
    const int half = lane >> 4, ml = lane & 15, kb8 = half * 8;
    const int qblk = blockIdx.x;
    const int bh = blockIdx.y;
    const int b = bh / H, h = bh % H, g = h / REP;

    const __bf16* Qp  = Q  + (size_t)b * T * D + (size_t)h * HD;         // stride D
    const __bf16* Kp  = Kb + (size_t)b * T * KV + (size_t)g * HD;        // stride KV
    const __bf16* Vtp = Vt + (size_t)(g * HD) * (2 * T) + (size_t)b * T; // stride B*T
    __bf16*       Op  = O  + (size_t)b * T * D + (size_t)h * HD;

    auto stageKV = [&](int buf, int k0) {
        for (int i = tid; i < 64 * 128 / 8; i += 128) {      // K block 64x128
            int r = i >> 4, c = (i & 15) * 8;
            async_ld_b128(&sK[buf][r][c], &Kp[(size_t)(k0 + r) * KV + c]);
        }
        for (int i = tid; i < 128 * 64 / 8; i += 128) {      // Vt block 128x64
            int r = i >> 3, c = (i & 7) * 8;
            async_ld_b128(&sVt[buf][r][c], &Vtp[(size_t)r * (2 * T) + k0 + c]);
        }
    };

    const int q0 = qblk * 64;
    for (int i = tid; i < 64 * 128 / 8; i += 128) {          // Q block 64x128
        int r = i >> 4, c = (i & 15) * 8;
        async_ld_b128(&sQ[r][c], &Qp[(size_t)(q0 + r) * D + c]);
    }
    stageKV(0, 0);

    float mrow[8], lrow[8];
    floatx8 oacc[8];
#pragma unroll
    for (int i = 0; i < 8; ++i) { mrow[i] = -1e30f; lrow[i] = 0.f; }
#pragma unroll
    for (int c = 0; c < 8; ++c) oacc[c] = (floatx8){};

    const int qrow_base = q0 + wave * 16;
    int buf = 0;

    for (int kblk = 0; kblk <= qblk; ++kblk) {
        const int k0 = kblk * 64;
        wait_async();
        __syncthreads();                                   // buffer ready everywhere
        if (kblk < qblk) stageKV(buf ^ 1, k0 + 64);        // prefetch under compute

        // S = Q K^T (16 rows x 64 keys per wave); 1/sqrt(HD) pre-folded into Q
        floatx8 s[4];
#pragma unroll
        for (int ct = 0; ct < 4; ++ct) s[ct] = (floatx8){};
#pragma unroll
        for (int kc = 0; kc < 4; ++kc) {
            int r = wave * 16 + ml;
            bf16x16 qf = frag_ld(&sQ[r][kc * 32 + kb8], &sQ[r][kc * 32 + kb8 + 16]);
#pragma unroll
            for (int ct = 0; ct < 4; ++ct) {
                int n = ct * 16 + ml;
                bf16x16 kf = frag_ld(&sK[buf][n][kc * 32 + kb8],
                                     &sK[buf][n][kc * 32 + kb8 + 16]);
                s[ct] = wmma_bf16(qf, kf, s[ct]);
            }
        }
        if (kblk == qblk) {   // causal mask only on the diagonal block
#pragma unroll
            for (int ct = 0; ct < 4; ++ct)
#pragma unroll
                for (int i = 0; i < 8; ++i) {
                    int qr = qrow_base + i + 8 * half;
                    int kc_ = k0 + ct * 16 + ml;
                    if (kc_ > qr) s[ct][i] = -1e30f;
                }
        }
        // online softmax (row = 16 lanes of one half-wave)
        float pm[8];
#pragma unroll
        for (int i = 0; i < 8; ++i)
            pm[i] = fmaxf(fmaxf(s[0][i], s[1][i]), fmaxf(s[2][i], s[3][i]));
#pragma unroll
        for (int off = 1; off < 16; off <<= 1)
#pragma unroll
            for (int i = 0; i < 8; ++i) pm[i] = fmaxf(pm[i], __shfl_xor(pm[i], off, 32));

        float fac[8], ps[8];
#pragma unroll
        for (int i = 0; i < 8; ++i) {
            float mn = fmaxf(mrow[i], pm[i]);
            fac[i] = __expf(mrow[i] - mn);
            mrow[i] = mn;
            ps[i] = 0.f;
        }
#pragma unroll
        for (int ct = 0; ct < 4; ++ct)
#pragma unroll
            for (int i = 0; i < 8; ++i) {
                float p = __expf(s[ct][i] - mrow[i]);
                s[ct][i] = p;
                ps[i] += p;
            }
#pragma unroll
        for (int off = 1; off < 16; off <<= 1)
#pragma unroll
            for (int i = 0; i < 8; ++i) ps[i] += __shfl_xor(ps[i], off, 32);
#pragma unroll
        for (int i = 0; i < 8; ++i) lrow[i] = lrow[i] * fac[i] + ps[i];
#pragma unroll
        for (int c = 0; c < 8; ++c)
#pragma unroll
            for (int i = 0; i < 8; ++i) oacc[c][i] *= fac[i];

        // P (C layout) -> per-wave LDS -> A-layout fragments
#pragma unroll
        for (int ct = 0; ct < 4; ++ct)
#pragma unroll
            for (int i = 0; i < 8; ++i)
                sP[wave][i + 8 * half][ct * 16 + ml] = (__bf16)s[ct][i];

#pragma unroll
        for (int kc = 0; kc < 2; ++kc) {
            bf16x16 pf = frag_ld(&sP[wave][ml][kc * 32 + kb8],
                                 &sP[wave][ml][kc * 32 + kb8 + 16]);
#pragma unroll
            for (int c = 0; c < 8; ++c) {
                int n = c * 16 + ml;
                bf16x16 vf = frag_ld(&sVt[buf][n][kc * 32 + kb8],
                                     &sVt[buf][n][kc * 32 + kb8 + 16]);
                oacc[c] = wmma_bf16(pf, vf, oacc[c]);
            }
        }
        buf ^= 1;
    }

    // normalize + store bf16 head slice
#pragma unroll
    for (int c = 0; c < 8; ++c)
#pragma unroll
        for (int i = 0; i < 8; ++i) {
            int qr = qrow_base + i + 8 * half;
            Op[(size_t)qr * D + c * 16 + ml] = (__bf16)(oacc[c][i] / lrow[i]);
        }
}

// ---------------------------------------------------------------- launch
extern "C" void kernel_launch(void* const* d_in, const int* in_sizes, int n_in,
                              void* d_out, int out_size, void* d_ws, size_t ws_size,
                              hipStream_t stream) {
    const int B = 2, T = 2048, D = 4096, H = 32, G = 8, HD = 128;
    const int M = B * T;           // 4096
    const int KVD = G * HD;        // 1024
    const float qscale = 0.08838834764831845f;   // 1/sqrt(128)

    const float* x  = (const float*)d_in[0];
    const float* Wq = (const float*)d_in[2];
    const float* bq = (const float*)d_in[3];
    const float* Wk = (const float*)d_in[4];
    const float* bk = (const float*)d_in[5];
    const float* Wv = (const float*)d_in[6];
    const float* bv = (const float*)d_in[7];
    const float* Wo = (const float*)d_in[8];
    const float* bo = (const float*)d_in[9];

    char* ws = (char*)d_ws;
    __bf16* xbf  = (__bf16*)(ws + 0);            // 32 MB  (reused as attn-out)
    __bf16* WqT  = (__bf16*)(ws + 33554432);     // 32 MB  [D][D]
    __bf16* WkT  = (__bf16*)(ws + 67108864);     //  8 MB  [KVD][D]
    __bf16* WvT  = (__bf16*)(ws + 75497472);     //  8 MB  [KVD][D]
    __bf16* WoT  = (__bf16*)(ws + 83886080);     // 32 MB  [D][D]
    __bf16* Qbf  = (__bf16*)(ws + 117440512);    // 32 MB  [M][D]
    __bf16* Kbf  = (__bf16*)(ws + 150994944);    //  8 MB  [M][KVD]
    __bf16* VtBf = (__bf16*)(ws + 159383552);    //  8 MB  [KVD][M] (transposed)
    __bf16* AObf = xbf;                          // alias: x dead after projections

    cast_f32_to_bf16<<<(M * D / 4 + 255) / 256, 256, 0, stream>>>(x, xbf, M * D);
    transpose_cast_kernel<<<dim3(D / 32, D / 32), 256, 0, stream>>>(Wq, WqT, D, D);
    transpose_cast_kernel<<<dim3(KVD / 32, D / 32), 256, 0, stream>>>(Wk, WkT, D, KVD);
    transpose_cast_kernel<<<dim3(KVD / 32, D / 32), 256, 0, stream>>>(Wv, WvT, D, KVD);
    transpose_cast_kernel<<<dim3(D / 32, D / 32), 256, 0, stream>>>(Wo, WoT, D, D);

    // Q/K projections (scale folded into Q); V projection stores V^T directly
    gemm_bf16_kernel<__bf16, false><<<dim3(D / GBN, M / GBM), 256, 0, stream>>>(
        xbf, WqT, bq, Qbf, M, D, D, qscale);
    gemm_bf16_kernel<__bf16, false><<<dim3(KVD / GBN, M / GBM), 256, 0, stream>>>(
        xbf, WkT, bk, Kbf, M, KVD, D, 1.0f);
    gemm_bf16_kernel<__bf16, true><<<dim3(KVD / GBN, M / GBM), 256, 0, stream>>>(
        xbf, WvT, bv, VtBf, M, KVD, D, 1.0f);

    // causal flash attention, one (b,h) x 64-row Q block per workgroup
    gqa_flash_kernel<<<dim3(T / 64, B * H), 128, 0, stream>>>(Qbf, Kbf, VtBf, AObf);

    // output projection -> f32
    gemm_bf16_kernel<float, false><<<dim3(D / GBN, M / GBM), 256, 0, stream>>>(
        AObf, WoT, bo, (float*)d_out, M, D, D, 1.0f);
}
